// Transformer_Encoder_Layer_31379031064639
// MI455X (gfx1250) — matrix operand order repaired and checked by reading, not compiled, not run on hardware
//
#include <hip/hip_runtime.h>
#include <stdint.h>

#define B_    4
#define L_    1024
#define HID_  768
#define NH_   12
#define D_    64
#define NTOK  (B_ * L_)          // 4096
#define SCALE_ 0.125f            // 1/sqrt(64)
#define NEG_  -1e15f

typedef __bf16 bf16_t;
typedef __bf16 v16bf __attribute__((ext_vector_type(16)));
typedef __bf16 v8bf  __attribute__((ext_vector_type(8)));
typedef float  v8f   __attribute__((ext_vector_type(8)));
typedef float  v4f   __attribute__((ext_vector_type(4)));

union BF16x16 { v16bf v; v8bf h[2]; };
union BF8U    { v8bf  v; uint4 u;  };

__device__ __forceinline__ bf16_t f2bf(float f) {
  uint32_t u = __builtin_bit_cast(uint32_t, f);
  uint32_t r = (u + 0x7fffu + ((u >> 16) & 1u)) >> 16;
  uint16_t s = (uint16_t)r;
  return __builtin_bit_cast(bf16_t, s);
}

// ---------------- elementwise f32 -> bf16 ----------------
__global__ __launch_bounds__(256) void cvt_f32_bf16(
    const float* __restrict__ in, bf16_t* __restrict__ out, int n) {
  int i = blockIdx.x * 256 + threadIdx.x;
  if (i < n) out[i] = f2bf(in[i]);
}

// ------------- W[k][n] (f32) -> Wt[n][k] (bf16) -------------
__global__ __launch_bounds__(256) void transpose_w(
    const float* __restrict__ w, bf16_t* __restrict__ wt) {
  int i = blockIdx.x * 256 + threadIdx.x;     // 0 .. 768*768-1
  int n = i / HID_, k = i % HID_;
  wt[i] = f2bf(w[(size_t)k * HID_ + n]);      // coalesced writes
}

// ---------------- bf16 GEMM:  out = A[M,768] x W[768,768] + bias ----------------
// Wt is pre-transposed [N=768][K=768] bf16 so the B operand is contiguous.
// One wave computes a 16(M) x 64(N) tile; K-loop is register double-buffered so
// loads for stage i+1 are in flight during the WMMAs of stage i.
// MODE 0: bf16 row-major   MODE 1: bf16 V-transposed [b,h,d,L]
// MODE 2: f32 row-major    MODE 3: relu -> bf16 row-major
template <int MODE>
__global__ __launch_bounds__(256) void gemm_bf16(
    const bf16_t* __restrict__ A, const bf16_t* __restrict__ Wt,
    const float* __restrict__ bias, void* __restrict__ outp) {
  const int lane = threadIdx.x & 31;
  const int wave = threadIdx.x >> 5;
  const int l16  = lane & 15;
  const int half = lane >> 4;
  const int tb = blockIdx.x * 32 + (wave & 1) * 16;   // token tile base
  const int nb = blockIdx.y * 256 + (wave >> 1) * 64; // feature tile base

  v8f zero = {0.f,0.f,0.f,0.f,0.f,0.f,0.f,0.f};
  v8f acc[4];
  #pragma unroll
  for (int t = 0; t < 4; ++t) acc[t] = zero;

  const bf16_t* arow = A  + (size_t)(tb + l16) * HID_ + half * 8;
  const bf16_t* brow = Wt + (size_t)(nb + l16) * HID_ + half * 8;

  BF16x16 abuf[2];
  BF16x16 bbuf[2][4];

  auto load_stage = [&](int kc, int buf) {
    abuf[buf].h[0] = *(const v8bf*)(arow + kc);        // A: K = kc+half*8..+7
    abuf[buf].h[1] = *(const v8bf*)(arow + kc + 16);   //    K = kc+16+half*8..+7
    #pragma unroll
    for (int t = 0; t < 4; ++t) {
      const bf16_t* bp = brow + (size_t)t * 16 * HID_ + kc;
      bbuf[buf][t].h[0] = *(const v8bf*)(bp);          // B: K = kc+half*8..+7
      bbuf[buf][t].h[1] = *(const v8bf*)(bp + 16);     //    K = kc+16+half*8..+7
    }
  };
  auto compute_stage = [&](int buf) {
    #pragma unroll
    for (int t = 0; t < 4; ++t)
      acc[t] = __builtin_amdgcn_wmma_f32_16x16x32_bf16(
          false, abuf[buf].v, false, bbuf[buf][t].v, (short)0, acc[t],
          false, false);
  };

  load_stage(0, 0);
  for (int kc = 0; kc < HID_; kc += 64) {
    load_stage(kc + 32, 1);                 // prefetch while computing buf0
    compute_stage(0);
    if (kc + 64 < HID_) load_stage(kc + 64, 0);
    compute_stage(1);
  }

  #pragma unroll
  for (int t = 0; t < 4; ++t) {
    const int n = nb + t * 16 + l16;
    const float bi = bias[n];
    #pragma unroll
    for (int r = 0; r < 8; ++r) {
      const int token = tb + half * 8 + r;   // C/D layout: M = half*8 + r
      float val = acc[t][r] + bi;
      if (MODE == 0) {
        ((bf16_t*)outp)[(size_t)token * HID_ + n] = f2bf(val);
      } else if (MODE == 1) {                // V transposed: [b,h,d,L]
        int hh = n >> 6, d = n & 63;
        int bb = token >> 10, l = token & 1023;
        ((bf16_t*)outp)[((size_t)(bb * NH_ + hh) * D_ + d) * L_ + l] = f2bf(val);
      } else if (MODE == 2) {
        ((float*)outp)[(size_t)token * HID_ + n] = val;
      } else {
        ((bf16_t*)outp)[(size_t)token * HID_ + n] = f2bf(fmaxf(val, 0.f));
      }
    }
  }
}

// ---------------- fused flash attention ----------------
// One wave handles 16 queries of one (b,h). Computes S^T = K_tile x Q^T so a
// lane pair owns one query column (softmax stats need only one shfl_xor(16)),
// adds SCALE*S + rel_pos (non-temporal: 201MB read-once stream must not evict
// L2-resident K/V/weights), masks only the boundary chunk, online softmax,
// then ctx^T += V^T x P^T.  Key loop is clamped to ceil(vlen/32) chunks:
// fully-masked chunks contribute exactly 0 (exp underflow), same as reference.
// vlen goes through readfirstlane so loop control is scalar (EXEC untouched,
// which WMMA requires anyway).
__global__ __launch_bounds__(128) void attn_fused(
    const bf16_t* __restrict__ q, const bf16_t* __restrict__ k,
    const bf16_t* __restrict__ vT, const float* __restrict__ rel,
    const int* __restrict__ seqlen, const int* __restrict__ lex,
    bf16_t* __restrict__ ctx) {
  __shared__ __attribute__((aligned(16))) bf16_t plds[4][16 * 32];

  const int lane = threadIdx.x & 31;
  const int wave = threadIdx.x >> 5;
  const int l16  = lane & 15;
  const int half = lane >> 4;

  const int idx = blockIdx.x * 4 + wave;  // 0..3071
  const int qt  = idx & 63;
  const int h   = (idx >> 6) % NH_;
  const int b   = idx / (64 * NH_);
  // uniform within the wave by construction -> make it an SGPR explicitly
  const int vlen = __builtin_amdgcn_readfirstlane(seqlen[b] + lex[0]);
  const int qtok = b * L_ + qt * 16 + l16;           // this lane's query token

  // Q as B operand (column = query, K-dim = d), two d-chunks of 32
  const bf16_t* qrow = q + (size_t)qtok * HID_ + h * D_;
  BF16x16 qB0, qB1;
  qB0.h[0] = *(const v8bf*)(qrow + half * 8);
  qB0.h[1] = *(const v8bf*)(qrow + 16 + half * 8);
  qB1.h[0] = *(const v8bf*)(qrow + 32 + half * 8);
  qB1.h[1] = *(const v8bf*)(qrow + 48 + half * 8);

  const float* relRow = rel + ((size_t)(b * NH_ + h) * L_ + (qt * 16 + l16)) * L_;
  const bf16_t* vbase = vT + (size_t)(b * NH_ + h) * D_ * L_;

  v8f zero = {0.f,0.f,0.f,0.f,0.f,0.f,0.f,0.f};
  v8f oacc[4] = {zero, zero, zero, zero};
  float mrun = -3.0e38f, srun = 0.f;
  bf16_t* prow = &plds[wave][l16 * 32];

  BF16x16 kbuf[2][2][2];      // [buf][key sub-tile][d-chunk]
  v4f     rbuf[2][2][2];      // [buf][key sub-tile][quad]

  auto load_k = [&](int kc, int buf) {
    #pragma unroll
    for (int si = 0; si < 2; ++si) {
      const bf16_t* krow =
          k + (size_t)(b * L_ + kc + si * 16 + l16) * HID_ + h * D_;
      kbuf[buf][si][0].h[0] = *(const v8bf*)(krow + half * 8);
      kbuf[buf][si][0].h[1] = *(const v8bf*)(krow + 16 + half * 8);
      kbuf[buf][si][1].h[0] = *(const v8bf*)(krow + 32 + half * 8);
      kbuf[buf][si][1].h[1] = *(const v8bf*)(krow + 48 + half * 8);
    }
  };
  auto load_rel = [&](int kc, int buf) {
    #pragma unroll
    for (int si = 0; si < 2; ++si) {
      const v4f* rp = (const v4f*)(relRow + kc + si * 16 + half * 8);
      rbuf[buf][si][0] = __builtin_nontemporal_load(rp);      // TH_NT stream
      rbuf[buf][si][1] = __builtin_nontemporal_load(rp + 1);
    }
  };

  auto process = [&](int kc, int buf) {
    BF16x16 va[4];                         // V^T tiles: issue loads first
    #pragma unroll
    for (int t = 0; t < 4; ++t) {
      const bf16_t* vrow = vbase + (size_t)(t * 16 + l16) * L_ + kc;
      va[t].h[0] = *(const v8bf*)(vrow + half * 8);
      va[t].h[1] = *(const v8bf*)(vrow + 16 + half * 8);
    }
    float sv[2][8];
    #pragma unroll
    for (int si = 0; si < 2; ++si) {
      v8f st = zero;
      st = __builtin_amdgcn_wmma_f32_16x16x32_bf16(
          false, kbuf[buf][si][0].v, false, qB0.v, (short)0, st, false, false);
      st = __builtin_amdgcn_wmma_f32_16x16x32_bf16(
          false, kbuf[buf][si][1].v, false, qB1.v, (short)0, st, false, false);
      v4f r0 = rbuf[buf][si][0], r1 = rbuf[buf][si][1];
      float rr[8] = {r0[0], r0[1], r0[2], r0[3], r1[0], r1[1], r1[2], r1[3]};
      #pragma unroll
      for (int r = 0; r < 8; ++r)
        sv[si][r] = st[r] * SCALE_ + rr[r];
    }
    if (kc + 32 > vlen) {                  // boundary chunk only (scalar branch)
      #pragma unroll
      for (int si = 0; si < 2; ++si)
        #pragma unroll
        for (int r = 0; r < 8; ++r) {
          int key = kc + si * 16 + half * 8 + r;  // D layout: M = half*8 + r
          if (key >= vlen) sv[si][r] = NEG_;
        }
    }
    // online softmax stats (lane pair = one query; combine halves once)
    float mc = -3.0e38f;
    #pragma unroll
    for (int si = 0; si < 2; ++si)
      #pragma unroll
      for (int r = 0; r < 8; ++r) mc = fmaxf(mc, sv[si][r]);
    mc = fmaxf(mc, __shfl_xor(mc, 16, 32));
    float mnew = fmaxf(mrun, mc);
    float corr = __expf(mrun - mnew);
    float ssum = 0.f;
    #pragma unroll
    for (int si = 0; si < 2; ++si)
      #pragma unroll
      for (int r = 0; r < 8; ++r) {
        float p = __expf(sv[si][r] - mnew);
        sv[si][r] = p;
        ssum += p;
      }
    ssum += __shfl_xor(ssum, 16, 32);
    srun = srun * corr + ssum;
    mrun = mnew;
    #pragma unroll
    for (int t = 0; t < 4; ++t)
      #pragma unroll
      for (int r = 0; r < 8; ++r) oacc[t][r] *= corr;

    // P -> LDS ([query][key 0..31] row-major bf16), reload as B operand
    #pragma unroll
    for (int si = 0; si < 2; ++si) {
      BF8U pk;
      #pragma unroll
      for (int r = 0; r < 8; ++r) pk.v[r] = f2bf(sv[si][r]);
      *reinterpret_cast<uint4*>(prow + si * 16 + half * 8) = pk.u;
    }
    asm volatile("" ::: "memory");   // same-wave DS ops are HW in-order
    BF16x16 pb;
    pb.h[0] = *(const v8bf*)(prow + half * 8);
    pb.h[1] = *(const v8bf*)(prow + 16 + half * 8);

    #pragma unroll
    for (int t = 0; t < 4; ++t)      // ctx^T += V^T(16d x 32key) x P^T
      oacc[t] = __builtin_amdgcn_wmma_f32_16x16x32_bf16(
          false, va[t].v, false, pb.v, (short)0, oacc[t], false, false);
    asm volatile("" ::: "memory");
  };

  // chunks beyond vlen contribute exactly 0 -> skip them entirely
  int kend = (vlen + 31) & ~31;
  if (kend > L_) kend = L_;
  const int nch = kend >> 5;                    // >= 1 (seq_len >= 1)

  load_k(0, 0);
  load_rel(0, 0);
  for (int i = 0; i < nch; i += 2) {
    const bool has1 = (i + 1 < nch);
    if (has1) { load_k((i + 1) * 32, 1); load_rel((i + 1) * 32, 1); }
    process(i * 32, 0);
    if (i + 2 < nch) { load_k((i + 2) * 32, 0); load_rel((i + 2) * 32, 0); }
    if (has1) process((i + 1) * 32, 1);
  }

  const float inv = 1.0f / srun;
  bf16_t* crow = ctx + (size_t)qtok * HID_ + h * D_;
  #pragma unroll
  for (int t = 0; t < 4; ++t) {
    BF8U pk;
    #pragma unroll
    for (int r = 0; r < 8; ++r) pk.v[r] = f2bf(oacc[t][r] * inv);
    *reinterpret_cast<uint4*>(crow + t * 16 + half * 8) = pk.u;
  }
}

// ---------------- LayerNorm of (y + y):  LN(2y)*g + b ----------------
// mode 0: bf16 out (feeds next GEMM)   mode 1: f32 out (final output)
__global__ __launch_bounds__(256) void ln_res(
    const float* __restrict__ y, const float* __restrict__ g,
    const float* __restrict__ bt, void* __restrict__ outp, int mode) {
  __shared__ float rs[256], rq[256];
  const int row = blockIdx.x, tid = threadIdx.x;
  const float* yr = y + (size_t)row * HID_;
  float t[3], ls = 0.f, lq = 0.f;
  #pragma unroll
  for (int j = 0; j < 3; ++j) {
    int i = tid + j * 256;
    float v = 2.f * yr[i];
    t[j] = v; ls += v; lq += v * v;
  }
  rs[tid] = ls; rq[tid] = lq;
  __syncthreads();
  for (int s = 128; s > 0; s >>= 1) {
    if (tid < s) { rs[tid] += rs[tid + s]; rq[tid] += rq[tid + s]; }
    __syncthreads();
  }
  const float mean = rs[0] * (1.f / HID_);
  const float var  = rq[0] * (1.f / HID_) - mean * mean;
  const float rinv = rsqrtf(var + 1e-5f);
  #pragma unroll
  for (int j = 0; j < 3; ++j) {
    int i = tid + j * 256;
    float o = (t[j] - mean) * rinv * g[i] + bt[i];
    if (mode == 0) ((bf16_t*)outp)[(size_t)row * HID_ + i] = f2bf(o);
    else           ((float*)outp)[(size_t)row * HID_ + i] = o;
  }
}

// ---------------- workspace layout (bytes) ----------------
static const size_t OFF_X   = 0;                                // x bf16  6.29MB
static const size_t OFF_WT  = OFF_X  + (size_t)NTOK * HID_ * 2; // 6 x Wt  7.08MB
static const size_t OFF_Q   = OFF_WT + (size_t)6 * HID_ * HID_ * 2;
static const size_t OFF_K   = OFF_Q  + (size_t)NTOK * HID_ * 2;
static const size_t OFF_VT  = OFF_K  + (size_t)NTOK * HID_ * 2;
static const size_t OFF_CTX = OFF_VT + (size_t)NTOK * HID_ * 2;
static const size_t OFF_Y   = OFF_CTX+ (size_t)NTOK * HID_ * 2; // f32    12.6MB
// reuse: h1 (LN out bf16) <- Q region, h2 (relu hidden bf16) <- K region,
//        FFN1 f32 out <- Y region.  Total ~51 MB.

extern "C" void kernel_launch(void* const* d_in, const int* in_sizes, int n_in,
                              void* d_out, int out_size, void* d_ws, size_t ws_size,
                              hipStream_t stream) {
  (void)in_sizes; (void)n_in; (void)out_size; (void)ws_size;
  const float* inp    = (const float*)d_in[0];
  const int*   seqlen = (const int*)  d_in[1];
  const int*   lex    = (const int*)  d_in[2];
  const float* rel    = (const float*)d_in[3];
  const float* Wsrc[6] = {(const float*)d_in[4],  (const float*)d_in[6],
                          (const float*)d_in[8],  (const float*)d_in[10],
                          (const float*)d_in[12], (const float*)d_in[14]};
  const float* bq = (const float*)d_in[5];
  const float* bk = (const float*)d_in[7];
  const float* bv = (const float*)d_in[9];
  const float* bo = (const float*)d_in[11];
  const float* b0 = (const float*)d_in[13];
  const float* b1 = (const float*)d_in[15];
  const float* lng = (const float*)d_in[16];
  const float* lnb = (const float*)d_in[17];

  char* ws = (char*)d_ws;
  bf16_t* xbf  = (bf16_t*)(ws + OFF_X);
  bf16_t* qb   = (bf16_t*)(ws + OFF_Q);
  bf16_t* kb   = (bf16_t*)(ws + OFF_K);
  bf16_t* vTb  = (bf16_t*)(ws + OFF_VT);
  bf16_t* ctxb = (bf16_t*)(ws + OFF_CTX);
  float*  yf   = (float*) (ws + OFF_Y);
  bf16_t* h1   = (bf16_t*)(ws + OFF_Q);   // reuse after attention
  bf16_t* h2   = (bf16_t*)(ws + OFF_K);   // reuse after attention

  cvt_f32_bf16<<<(NTOK * HID_) / 256, 256, 0, stream>>>(inp, xbf, NTOK * HID_);
  for (int i = 0; i < 6; ++i) {
    bf16_t* wt = (bf16_t*)(ws + OFF_WT + (size_t)i * HID_ * HID_ * 2);
    transpose_w<<<(HID_ * HID_) / 256, 256, 0, stream>>>(Wsrc[i], wt);
  }
  bf16_t* wtQ = (bf16_t*)(ws + OFF_WT + (size_t)0 * HID_ * HID_ * 2);
  bf16_t* wtK = (bf16_t*)(ws + OFF_WT + (size_t)1 * HID_ * HID_ * 2);
  bf16_t* wtV = (bf16_t*)(ws + OFF_WT + (size_t)2 * HID_ * HID_ * 2);
  bf16_t* wtO = (bf16_t*)(ws + OFF_WT + (size_t)3 * HID_ * HID_ * 2);
  bf16_t* wt0 = (bf16_t*)(ws + OFF_WT + (size_t)4 * HID_ * HID_ * 2);
  bf16_t* wt1 = (bf16_t*)(ws + OFF_WT + (size_t)5 * HID_ * HID_ * 2);

  dim3 gg(NTOK / 32, HID_ / 256);   // (128, 3), 256 threads = 8 waves
  gemm_bf16<0><<<gg, 256, 0, stream>>>(xbf, wtQ, bq, qb);
  gemm_bf16<0><<<gg, 256, 0, stream>>>(xbf, wtK, bk, kb);
  gemm_bf16<1><<<gg, 256, 0, stream>>>(xbf, wtV, bv, vTb);

  attn_fused<<<(B_ * NH_ * (L_ / 16)) / 4, 128, 0, stream>>>(
      qb, kb, vTb, rel, seqlen, lex, ctxb);

  gemm_bf16<2><<<gg, 256, 0, stream>>>(ctxb, wtO, bo, yf);
  ln_res<<<NTOK, 256, 0, stream>>>(yf, lng, lnb, h1, 0);
  gemm_bf16<3><<<gg, 256, 0, stream>>>(h1, wt0, b0, h2);
  gemm_bf16<2><<<gg, 256, 0, stream>>>(h2, wt1, b1, yf);
  ln_res<<<NTOK, 256, 0, stream>>>(yf, lng, lnb, d_out, 1);
}
